// MultiHeadCrossAttention_1795296330349
// MI455X (gfx1250) — matrix operand order repaired and checked by reading, not compiled, not run on hardware
//
#include <hip/hip_runtime.h>
#include <hip/hip_bf16.h>
#include <cmath>

typedef __bf16 bf16_t;
typedef __attribute__((ext_vector_type(16))) __bf16 v16bf;
typedef __attribute__((ext_vector_type(8)))  __bf16 v8bf;
typedef __attribute__((ext_vector_type(8)))  float  v8f;

#define DEV static __device__ __forceinline__

// ---------------------------------------------------------------- constants
#define D_MODEL 1152
#define NUM_H   16
#define HEAD_D  72
#define BATCH   4
#define SEQ_N   4096
#define SEQ_M   300
#define DPAD    96     // head dim padded for K-loops (3 chunks of 32)
#define MPAD_K  304    // m padded to 19 tiles of 16 for QK^T columns
#define MPAD_V  320    // m padded to 10 chunks of 32 for PV K-loop
#define DPAD_V  80     // head dim padded to 5 tiles of 16 for PV columns

// workspace layout (bytes)
#define SZ_WQT   ((size_t)D_MODEL * D_MODEL * 2)          // Wq^T  bf16
#define SZ_WKVT  ((size_t)D_MODEL * 2 * D_MODEL * 2)      // Wkv^T bf16
#define SZ_WPT   ((size_t)D_MODEL * D_MODEL * 2)          // Wp^T  bf16
#define SZ_QPAD  ((size_t)BATCH * NUM_H * SEQ_N * DPAD * 2)
#define SZ_KPAD  ((size_t)BATCH * NUM_H * MPAD_K * DPAD * 2)
#define SZ_VT    ((size_t)BATCH * NUM_H * DPAD_V * MPAD_V * 2)
#define SZ_AO    ((size_t)BATCH * SEQ_N * D_MODEL * 2)
#define OFF_WQT  ((size_t)0)
#define OFF_WKVT (OFF_WQT + SZ_WQT)
#define OFF_WPT  (OFF_WKVT + SZ_WKVT)
#define OFF_QPAD (OFF_WPT + SZ_WPT)
#define OFF_KPAD (OFF_QPAD + SZ_QPAD)
#define OFF_VT   (OFF_KPAD + SZ_KPAD)
#define OFF_AO   (OFF_VT + SZ_VT)
#define WS_TOTAL (OFF_AO + SZ_AO)

// ---------------------------------------------------------------- WMMA helpers
DEV v8f wmma_bf16(v16bf a, v16bf b, v8f c) {
  // D = A(16x32) * B(32x16) + C, f32 accumulate
  return __builtin_amdgcn_wmma_f32_16x16x32_bf16(false, a, false, b, (short)0, c,
                                                 false, false);
}

// A fragment (16x32, bf16) from row-major f32 source.
// ISA layout: lane holds row = lane%16; K = (lane/16)*8 + {0..7} and +16.
DEV v16bf load_a_f32(const float* __restrict__ base, int ld, int row0, int kc, int lane) {
  const float* p = base + (size_t)(row0 + (lane & 15)) * ld + kc + ((lane >> 4) << 3);
  float4 f0 = *(const float4*)(p + 0);
  float4 f1 = *(const float4*)(p + 4);
  float4 f2 = *(const float4*)(p + 16);
  float4 f3 = *(const float4*)(p + 20);
  v16bf a;
  a[0]=(bf16_t)f0.x;  a[1]=(bf16_t)f0.y;  a[2]=(bf16_t)f0.z;  a[3]=(bf16_t)f0.w;
  a[4]=(bf16_t)f1.x;  a[5]=(bf16_t)f1.y;  a[6]=(bf16_t)f1.z;  a[7]=(bf16_t)f1.w;
  a[8]=(bf16_t)f2.x;  a[9]=(bf16_t)f2.y;  a[10]=(bf16_t)f2.z; a[11]=(bf16_t)f2.w;
  a[12]=(bf16_t)f3.x; a[13]=(bf16_t)f3.y; a[14]=(bf16_t)f3.z; a[15]=(bf16_t)f3.w;
  return a;
}

// A fragment from row-major bf16 source (global or LDS).
DEV v16bf load_a_bf16(const bf16_t* __restrict__ base, int ld, int row0, int kc, int lane) {
  const bf16_t* p = base + (size_t)(row0 + (lane & 15)) * ld + kc + ((lane >> 4) << 3);
  v8bf lo = *(const v8bf*)(p);
  v8bf hi = *(const v8bf*)(p + 16);
  return __builtin_shufflevector(lo, hi, 0,1,2,3,4,5,6,7,8,9,10,11,12,13,14,15);
}

// B fragment (32x16, bf16) from K-contiguous ("transposed") bf16 source:
// row `col` of the source holds that output-column's K values contiguously.
// ISA layout: lane holds col = lane%16; K = (lane/16)*8 + {0..7} and +16.
DEV v16bf load_b_bf16(const bf16_t* __restrict__ base, int ld, int col0, int kc, int lane) {
  const bf16_t* p = base + (size_t)(col0 + (lane & 15)) * ld + kc + ((lane >> 4) << 3);
  v8bf lo = *(const v8bf*)(p);
  v8bf hi = *(const v8bf*)(p + 16);
  return __builtin_shufflevector(lo, hi, 0,1,2,3,4,5,6,7,8,9,10,11,12,13,14,15);
}

// ---------------------------------------------------------------- transpose f32 -> bf16
__global__ void __launch_bounds__(256) transpose_f32_to_bf16(
    const float* __restrict__ in, bf16_t* __restrict__ out, int rows, int cols) {
  __shared__ float tile[32][33];
  const int tx = threadIdx.x, ty = threadIdx.y;
  const int cbase = blockIdx.x * 32, rbase = blockIdx.y * 32;
  #pragma unroll
  for (int j = 0; j < 32; j += 8) {
    const int r = rbase + ty + j, c = cbase + tx;
    if (r < rows && c < cols) tile[ty + j][tx] = in[(size_t)r * cols + c];
  }
  __syncthreads();
  #pragma unroll
  for (int j = 0; j < 32; j += 8) {
    const int r = rbase + tx, c = cbase + ty + j;
    if (r < rows && c < cols) out[(size_t)c * rows + r] = (bf16_t)tile[tx][ty + j];
  }
}

// ---------------------------------------------------------------- q projection
// q = (x @ Wq + bq) * scale  -> q_pad[b,h,n,DPAD] bf16
__global__ void __launch_bounds__(256) qproj_kernel(
    const float* __restrict__ x, const bf16_t* __restrict__ wqt,
    const float* __restrict__ bq, bf16_t* __restrict__ q_pad) {
  const int lane = threadIdx.x & 31;
  const int wave = threadIdx.x >> 5;
  const int m0 = blockIdx.x * 128 + (wave & 3) * 32;
  const int n0 = blockIdx.y * 64 + (wave >> 2) * 32;
  v8f acc[2][2] = {};
  for (int kc = 0; kc < D_MODEL; kc += 32) {
    v16bf a0 = load_a_f32(x, D_MODEL, m0, kc, lane);
    v16bf a1 = load_a_f32(x, D_MODEL, m0 + 16, kc, lane);
    v16bf b0 = load_b_bf16(wqt, D_MODEL, n0, kc, lane);
    v16bf b1 = load_b_bf16(wqt, D_MODEL, n0 + 16, kc, lane);
    acc[0][0] = wmma_bf16(a0, b0, acc[0][0]);
    acc[0][1] = wmma_bf16(a0, b1, acc[0][1]);
    acc[1][0] = wmma_bf16(a1, b0, acc[1][0]);
    acc[1][1] = wmma_bf16(a1, b1, acc[1][1]);
  }
  const float scale = 0.11785113019775793f; // 1/sqrt(72)
  #pragma unroll
  for (int j = 0; j < 2; ++j) {
    const int col = n0 + j * 16 + (lane & 15);
    const int h = col / HEAD_D, d = col % HEAD_D;
    const float bias = bq[col];
    #pragma unroll
    for (int i = 0; i < 2; ++i) {
      #pragma unroll
      for (int c = 0; c < 8; ++c) {
        const int r = m0 + i * 16 + ((lane >> 4) << 3) + c; // global row in [0,16384)
        const int b = r >> 12, n = r & (SEQ_N - 1);
        const float v = (acc[i][j][c] + bias) * scale;
        q_pad[((size_t)(b * NUM_H + h) * SEQ_N + n) * DPAD + d] = (bf16_t)v;
      }
    }
  }
}

// ---------------------------------------------------------------- kv projection
// kv = cond @ Wkv + bkv; k -> k_pad[b,h,MPAD_K,DPAD]; v -> v_t[b,h,DPAD_V,MPAD_V]
__global__ void __launch_bounds__(128) kvproj_kernel(
    const float* __restrict__ cond, const bf16_t* __restrict__ wkvt,
    const float* __restrict__ bkv, bf16_t* __restrict__ k_pad,
    bf16_t* __restrict__ v_t) {
  const int lane = threadIdx.x & 31;
  const int wave = threadIdx.x >> 5;
  const int m0 = blockIdx.x * 16;                 // rows: B*M = 1200 = 75 tiles
  const int n0 = blockIdx.y * 128 + wave * 32;    // cols: 2304
  v8f acc[2] = {};
  for (int kc = 0; kc < D_MODEL; kc += 32) {
    v16bf a  = load_a_f32(cond, D_MODEL, m0, kc, lane);
    v16bf b0 = load_b_bf16(wkvt, D_MODEL, n0, kc, lane);
    v16bf b1 = load_b_bf16(wkvt, D_MODEL, n0 + 16, kc, lane);
    acc[0] = wmma_bf16(a, b0, acc[0]);
    acc[1] = wmma_bf16(a, b1, acc[1]);
  }
  #pragma unroll
  for (int j = 0; j < 2; ++j) {
    const int col = n0 + j * 16 + (lane & 15);
    const float bias = bkv[col];
    #pragma unroll
    for (int c = 0; c < 8; ++c) {
      const int r = m0 + ((lane >> 4) << 3) + c;  // in [0,1200)
      const int b = r / SEQ_M, m = r % SEQ_M;
      const float v = acc[j][c] + bias;
      if (col < D_MODEL) {                         // k part
        const int h = col / HEAD_D, d = col % HEAD_D;
        k_pad[((size_t)(b * NUM_H + h) * MPAD_K + m) * DPAD + d] = (bf16_t)v;
      } else {                                     // v part (stored transposed)
        const int jj = col - D_MODEL;
        const int h = jj / HEAD_D, d = jj % HEAD_D;
        v_t[((size_t)(b * NUM_H + h) * DPAD_V + d) * MPAD_V + m] = (bf16_t)v;
      }
    }
  }
}

// ---------------------------------------------------------------- attention
// Per wave: one 16-row q tile for one (b,h). Logits over all m, softmax, PV.
__global__ void __launch_bounds__(128) attn_kernel(
    const bf16_t* __restrict__ q_pad, const bf16_t* __restrict__ k_pad,
    const bf16_t* __restrict__ v_t, const int* __restrict__ mask,
    bf16_t* __restrict__ attn_out) {
  __shared__ __attribute__((aligned(64))) bf16_t pbuf[4][16][MPAD_V]; // 40 KB
  const int lane = threadIdx.x & 31;
  const int wave = threadIdx.x >> 5;
  const int bh = blockIdx.y;
  const int b = bh >> 4, h = bh & 15;
  const int nbase = blockIdx.x * 64 + wave * 16;
  const bf16_t* qrows = q_pad + ((size_t)bh * SEQ_N + nbase) * DPAD;
  const bf16_t* krows = k_pad + (size_t)bh * MPAD_K * DPAD;
  const bf16_t* vrows = v_t + (size_t)bh * DPAD_V * MPAD_V;
  const int maskb = mask[b];
  const int mlo = lane & 15;
  const int rloc = (lane >> 4) << 3;

  // S[t]: 16x16 logits tile for m-tile t (19 tiles cover MPAD_K=304)
  v8f S[19] = {};
  #pragma unroll
  for (int kc = 0; kc < DPAD; kc += 32) {
    v16bf a = load_a_bf16(qrows, DPAD, 0, kc, lane);
    #pragma unroll
    for (int t = 0; t < 19; ++t) {
      v16bf bfr = load_b_bf16(krows, DPAD, t * 16, kc, lane);
      S[t] = wmma_bf16(a, bfr, S[t]);
    }
  }

  // softmax per row (row group = 16 lanes sharing lane>>4), with mask-add (+1 if m<mask[b])
  #pragma unroll
  for (int c = 0; c < 8; ++c) {
    float mx = -3.0e38f;
    #pragma unroll
    for (int t = 0; t < 19; ++t) {
      const int m = t * 16 + mlo;
      float v = S[t][c];
      v = (m < SEQ_M) ? (v + ((m < maskb) ? 1.0f : 0.0f)) : -3.0e38f;
      mx = fmaxf(mx, v);
    }
    #pragma unroll
    for (int off = 1; off < 16; off <<= 1) mx = fmaxf(mx, __shfl_xor(mx, off, 32));
    float sum = 0.0f;
    #pragma unroll
    for (int t = 0; t < 19; ++t) {
      const int m = t * 16 + mlo;
      float v = S[t][c];
      v = v + ((m < maskb) ? 1.0f : 0.0f);
      const float e = (m < SEQ_M) ? __expf(v - mx) : 0.0f;
      S[t][c] = e;
      sum += e;
    }
    #pragma unroll
    for (int off = 1; off < 16; off <<= 1) sum += __shfl_xor(sum, off, 32);
    const float inv = 1.0f / sum;
    const int r = rloc + c;
    #pragma unroll
    for (int t = 0; t < 19; ++t)
      pbuf[wave][r][t * 16 + mlo] = (bf16_t)(S[t][c] * inv);
  }
  // zero-fill P columns 304..319 (each lane writes 8)
  {
    const int r = lane >> 1;
    const int cb = SEQ_M + 4 + (lane & 1) * 8; // 304 or 312
    #pragma unroll
    for (int i = 0; i < 8; ++i) pbuf[wave][r][cb + i] = (bf16_t)0.0f;
  }
  __syncthreads();

  // O = P (16 x MPAD_V) @ v_t^T -> 16 x DPAD_V
  v8f O[5] = {};
  const bf16_t* prows = &pbuf[wave][0][0];
  #pragma unroll
  for (int kc = 0; kc < MPAD_V; kc += 32) {
    v16bf a = load_a_bf16(prows, MPAD_V, 0, kc, lane);
    #pragma unroll
    for (int dt = 0; dt < 5; ++dt) {
      v16bf bfr = load_b_bf16(vrows, MPAD_V, dt * 16, kc, lane);
      O[dt] = wmma_bf16(a, bfr, O[dt]);
    }
  }
  #pragma unroll
  for (int dt = 0; dt < 5; ++dt) {
    const int d = dt * 16 + mlo;
    if (d < HEAD_D) {
      #pragma unroll
      for (int c = 0; c < 8; ++c) {
        const int n = nbase + rloc + c;
        attn_out[((size_t)(b * SEQ_N + n)) * D_MODEL + h * HEAD_D + d] = (bf16_t)O[dt][c];
      }
    }
  }
}

// ---------------------------------------------------------------- output projection
__global__ void __launch_bounds__(256) oproj_kernel(
    const bf16_t* __restrict__ ao, const bf16_t* __restrict__ wpt,
    const float* __restrict__ bp, float* __restrict__ out) {
  const int lane = threadIdx.x & 31;
  const int wave = threadIdx.x >> 5;
  const int m0 = blockIdx.x * 128 + (wave & 3) * 32;
  const int n0 = blockIdx.y * 64 + (wave >> 2) * 32;
  v8f acc[2][2] = {};
  for (int kc = 0; kc < D_MODEL; kc += 32) {
    v16bf a0 = load_a_bf16(ao, D_MODEL, m0, kc, lane);
    v16bf a1 = load_a_bf16(ao, D_MODEL, m0 + 16, kc, lane);
    v16bf b0 = load_b_bf16(wpt, D_MODEL, n0, kc, lane);
    v16bf b1 = load_b_bf16(wpt, D_MODEL, n0 + 16, kc, lane);
    acc[0][0] = wmma_bf16(a0, b0, acc[0][0]);
    acc[0][1] = wmma_bf16(a0, b1, acc[0][1]);
    acc[1][0] = wmma_bf16(a1, b0, acc[1][0]);
    acc[1][1] = wmma_bf16(a1, b1, acc[1][1]);
  }
  #pragma unroll
  for (int j = 0; j < 2; ++j) {
    const int col = n0 + j * 16 + (lane & 15);
    const float bias = bp[col];
    #pragma unroll
    for (int i = 0; i < 2; ++i) {
      #pragma unroll
      for (int c = 0; c < 8; ++c) {
        const int r = m0 + i * 16 + ((lane >> 4) << 3) + c;
        out[(size_t)r * D_MODEL + col] = acc[i][j][c] + bias;
      }
    }
  }
}

// ---------------------------------------------------------------- launcher
extern "C" void kernel_launch(void* const* d_in, const int* in_sizes, int n_in,
                              void* d_out, int out_size, void* d_ws, size_t ws_size,
                              hipStream_t stream) {
  (void)in_sizes; (void)n_in; (void)out_size;
  if (ws_size < WS_TOTAL) return; // not enough scratch; bail deterministically

  const float* x    = (const float*)d_in[0];
  const float* cond = (const float*)d_in[1];
  const int*   mask = (const int*)d_in[2];
  const float* Wq   = (const float*)d_in[3];
  const float* bq   = (const float*)d_in[4];
  const float* Wkv  = (const float*)d_in[5];
  const float* bkv  = (const float*)d_in[6];
  const float* Wp   = (const float*)d_in[7];
  const float* bp   = (const float*)d_in[8];
  float* out = (float*)d_out;

  char* ws = (char*)d_ws;
  bf16_t* wqt   = (bf16_t*)(ws + OFF_WQT);
  bf16_t* wkvt  = (bf16_t*)(ws + OFF_WKVT);
  bf16_t* wpt   = (bf16_t*)(ws + OFF_WPT);
  bf16_t* q_pad = (bf16_t*)(ws + OFF_QPAD);
  bf16_t* k_pad = (bf16_t*)(ws + OFF_KPAD);
  bf16_t* v_t   = (bf16_t*)(ws + OFF_VT);
  bf16_t* ao    = (bf16_t*)(ws + OFF_AO);

  // zero the padded q/k/v region (contiguous) so pad lanes are exact zeros
  hipMemsetAsync(ws + OFF_QPAD, 0, SZ_QPAD + SZ_KPAD + SZ_VT, stream);

  dim3 tb(32, 8);
  transpose_f32_to_bf16<<<dim3(D_MODEL / 32, D_MODEL / 32), tb, 0, stream>>>(
      Wq, wqt, D_MODEL, D_MODEL);
  transpose_f32_to_bf16<<<dim3(2 * D_MODEL / 32, D_MODEL / 32), tb, 0, stream>>>(
      Wkv, wkvt, D_MODEL, 2 * D_MODEL);
  transpose_f32_to_bf16<<<dim3(D_MODEL / 32, D_MODEL / 32), tb, 0, stream>>>(
      Wp, wpt, D_MODEL, D_MODEL);

  qproj_kernel<<<dim3((BATCH * SEQ_N) / 128, D_MODEL / 64), 256, 0, stream>>>(
      x, wqt, bq, q_pad);
  kvproj_kernel<<<dim3((BATCH * SEQ_M) / 16, (2 * D_MODEL) / 128), 128, 0, stream>>>(
      cond, wkvt, bkv, k_pad, v_t);
  attn_kernel<<<dim3(SEQ_N / 64, BATCH * NUM_H), 128, 0, stream>>>(
      q_pad, k_pad, v_t, mask, ao);
  oproj_kernel<<<dim3((BATCH * SEQ_N) / 128, D_MODEL / 64), 256, 0, stream>>>(
      ao, wpt, bp, out);
}
